// DilatedAttention_31507880083802
// MI455X (gfx1250) — matrix operand order repaired and checked by reading, not compiled, not run on hardware
//
#include <hip/hip_runtime.h>
#include <hip/hip_bf16.h>
#include <type_traits>

typedef __attribute__((ext_vector_type(16))) _Float16 v16h;
typedef __attribute__((ext_vector_type(8)))  float    v8f;

union AFrag { v16h v; _Float16 e[16]; uint4 q[2]; };
union CFrag { v8f  v; float    e[8];  };

#define SEQ   8192
#define NH    12
#define HD    64
#define MLEN  2048
#define NOUT  (2*SEQ*NH*HD)

// DPP16 lane permute (VALU, no LDS traffic). All lanes active (EXEC all-1s in
// this kernel), perms stay within their 16-lane row => masks 0xF/0xF are safe.
template<int CTRL>
__device__ __forceinline__ float dppf(float x) {
  return __builtin_bit_cast(float,
      __builtin_amdgcn_mov_dpp(__builtin_bit_cast(int, x), CTRL, 0xF, 0xF, false));
}
// Butterfly reduction over each 16-lane group: xor1, xor2, ~xor4, ~xor8.
__device__ __forceinline__ float rmax16(float x) {
  x = fmaxf(x, dppf<0xB1>(x));    // quad_perm [1,0,3,2]  (xor 1)
  x = fmaxf(x, dppf<0x4E>(x));    // quad_perm [2,3,0,1]  (xor 2)
  x = fmaxf(x, dppf<0x141>(x));   // row_half_mirror      (~xor 4)
  x = fmaxf(x, dppf<0x140>(x));   // row_mirror           (~xor 8)
  return x;
}
__device__ __forceinline__ float rsum16(float x) {
  x += dppf<0xB1>(x);
  x += dppf<0x4E>(x);
  x += dppf<0x141>(x);
  x += dppf<0x140>(x);
  return x;
}

__global__ void da_zero(float* __restrict__ out) {
  int i = blockIdx.x * 256 + threadIdx.x;
  if (i < NOUT) out[i] = 0.0f;
}

__global__ __launch_bounds__(128)
void da_flash(const float* __restrict__ Q, const float* __restrict__ Kp,
              const float* __restrict__ Vp, const int* __restrict__ causalPtr,
              float* __restrict__ Out) {
  __shared__ _Float16 kTile[2][32][72];   // [buf][key][feature], row-major
  __shared__ _Float16 vT[2][64][40];      // [buf][feature][key], transposed
  __shared__ _Float16 pT[4][16][40];      // per-wave P staging [row][col]

  const int causal = causalPtr[0];

  // ---- decode problem: 56 = 32(g0) + 16(g1) + 8(g2) ----
  int p = blockIdx.y;
  int grp, b, seg, hh;
  if (p < 32)      { grp = 0; int q = p;      b = q >> 4; q &= 15; seg = q >> 2; hh = (q & 3); }
  else if (p < 48) { grp = 1; int q = p - 32; b = q >> 3; q &= 7;  seg = q >> 2; hh = (q & 3) + 4; }
  else             { grp = 2; int q = p - 48; b = q >> 2; seg = 0;               hh = (q & 3) + 8; }
  const int rr  = (grp == 0) ? 1 : ((grp == 1) ? 2 : 4);
  const int ss  = (grp == 0) ? 2048 : ((grp == 1) ? 4096 : 8192);
  const int off = grp;                                 // i % r = {0,1,2}
  const size_t base = ((size_t)(b * SEQ + seg * ss + off) * NH + hh) * HD;
  const int rowStride = rr * NH * HD;                  // stride between dilated rows

  const int tid  = threadIdx.x;
  // wave index is uniform across the wave: pin it to an SGPR so the tile
  // classification lowers to scalar branches (s_cbranch) instead of saveexec,
  // keeping EXEC structurally all-1s around the WMMAs.
  const int wave = __builtin_amdgcn_readfirstlane(tid >> 5);
  const int lane = tid & 31;
  const int n16  = lane & 15;
  const int c    = lane >> 4;
  const int qb   = blockIdx.x * 64;                    // 64 query rows per block

  // staging thread mappings
  const int ksrow = tid >> 2;              // K: key row 0..31
  const int ksfb  = (tid & 3) << 4;        // K: feature base {0,16,32,48}
  const int vsft  = tid & 63;              // V: feature 0..63
  const int vskb  = (tid >> 6) << 4;       // V: key base {0,16}

  // ---- Q A-fragments (f32 -> f16), scale*log2(e) folded in, loaded once ----
  const float QSCALE = 0.125f * 1.44269504088896340736f;  // 1/sqrt(64) * log2(e)
  AFrag qa[2];
  {
    const float* qp = Q + base + (size_t)(qb + wave * 16 + n16) * rowStride;
    #pragma unroll
    for (int ks = 0; ks < 2; ks++)
      #pragma unroll
      for (int i = 0; i < 16; i++) {
        int feat = 32 * ks + 16 * c + i;
        qa[ks].e[i] = (_Float16)(qp[feat] * QSCALE);
      }
  }

  CFrag o[4];
  #pragma unroll
  for (int f = 0; f < 4; f++)
    #pragma unroll
    for (int r = 0; r < 8; r++) o[f].e[r] = 0.0f;
  float mrow[8], lrow[8];
  #pragma unroll
  for (int r = 0; r < 8; r++) { mrow[r] = -1e30f; lrow[r] = 0.0f; }

  const int ktmax = causal ? ((qb + 63) >> 5) : (MLEN / 32 - 1);
  const int qmin  = qb + 16 * wave;        // wave-uniform (SGPR) row range
  const int qmax  = qmin + 15;

  // ---- incrementally-advanced prefetch pointers (no 64-bit muls in loop) ----
  const size_t tileStep = (size_t)32 * rowStride;                 // floats per key tile
  const float* kPre = Kp + base + (size_t)ksrow * rowStride + ksfb;
  const float* vPre = Vp + base + (size_t)vskb * rowStride + vsft;

  // ---- prologue: fetch tile 0 into registers, stage to buffer 0 ----
  AFrag kpre, vpre;
  {
    #pragma unroll
    for (int u = 0; u < 16; u++) kpre.e[u] = (_Float16)kPre[u];
    #pragma unroll
    for (int u = 0; u < 16; u++) vpre.e[u] = (_Float16)vPre[(size_t)u * rowStride];
    uint4* kd = (uint4*)&kTile[0][ksrow][ksfb];
    kd[0] = kpre.q[0]; kd[1] = kpre.q[1];
    uint4* vd = (uint4*)&vT[0][vsft][vskb];
    vd[0] = vpre.q[0]; vd[1] = vpre.q[1];
  }
  if (ktmax > 0) { kPre += tileStep; vPre += tileStep; }          // now at tile 1

  // ---- specialized tile compute: DOMASK selected at compile time ----
  auto tileCompute = [&](int kbase, int cur, auto domask_c) {
    constexpr bool DOMASK = decltype(domask_c)::value;
    CFrag s0, s1;
    #pragma unroll
    for (int r = 0; r < 8; r++) { s0.e[r] = 0.0f; s1.e[r] = 0.0f; }
    {
      AFrag kbl, kbh;
      const _Float16* kr = &kTile[cur][n16][0];
      #pragma unroll
      for (int i = 0; i < 16; i++) { kbl.e[i] = kr[16 * c + i]; kbh.e[i] = kr[32 + 16 * c + i]; }
      s0.v = __builtin_amdgcn_wmma_f32_16x16x32_f16(false, qa[0].v, false, kbl.v, (short)0, s0.v, false, false);
      s0.v = __builtin_amdgcn_wmma_f32_16x16x32_f16(false, qa[1].v, false, kbh.v, (short)0, s0.v, false, false);
    }
    {
      AFrag kbl, kbh;
      const _Float16* kr = &kTile[cur][16 + n16][0];
      #pragma unroll
      for (int i = 0; i < 16; i++) { kbl.e[i] = kr[16 * c + i]; kbh.e[i] = kr[32 + 16 * c + i]; }
      s1.v = __builtin_amdgcn_wmma_f32_16x16x32_f16(false, qa[0].v, false, kbl.v, (short)0, s1.v, false, false);
      s1.v = __builtin_amdgcn_wmma_f32_16x16x32_f16(false, qa[1].v, false, kbh.v, (short)0, s1.v, false, false);
    }

    // online softmax in exp2 domain (D-frag row = r + 8c)
    #pragma unroll
    for (int r = 0; r < 8; r++) {
      float v0 = s0.e[r];
      float v1 = s1.e[r];
      if (DOMASK) {
        int qm = qmin + r + 8 * c;
        v0 = (kbase + n16      > qm) ? -1e30f : v0;
        v1 = (kbase + 16 + n16 > qm) ? -1e30f : v1;
      }
      float tmax = rmax16(fmaxf(v0, v1));
      float mnew = fmaxf(mrow[r], tmax);
      float corr = __builtin_amdgcn_exp2f(mrow[r] - mnew);
      float p0 = __builtin_amdgcn_exp2f(v0 - mnew);
      float p1 = __builtin_amdgcn_exp2f(v1 - mnew);
      float rs = rsum16(p0 + p1);
      lrow[r] = lrow[r] * corr + rs;
      mrow[r] = mnew;
      #pragma unroll
      for (int f = 0; f < 4; f++) o[f].e[r] *= corr;
      pT[wave][r + 8 * c][n16]      = (_Float16)p0;
      pT[wave][r + 8 * c][16 + n16] = (_Float16)p1;
    }
    // per-wave DS RAW fence (DS ops are in-order per wave) + compiler barrier
    asm volatile("s_wait_dscnt 0" ::: "memory");

    // P D-layout -> A-layout via LDS
    AFrag pa;
    {
      const _Float16* pr = &pT[wave][n16][0];
      #pragma unroll
      for (int i = 0; i < 16; i++)
        pa.e[i] = pr[16 * c + i];
    }
    // O += P V (4 output feature blocks)
    #pragma unroll
    for (int f = 0; f < 4; f++) {
      AFrag vb;
      const _Float16* vr = &vT[cur][16 * f + n16][0];
      #pragma unroll
      for (int i = 0; i < 16; i++) vb.e[i] = vr[16 * c + i];
      o[f].v = __builtin_amdgcn_wmma_f32_16x16x32_f16(false, pa.v, false, vb.v, (short)0, o[f].v, false, false);
    }
  };

  for (int kt = 0; kt <= ktmax; kt++) {
    const int kbase = kt * 32;
    const int cur = kt & 1;
    __syncthreads();   // staged tile `cur` now visible to all waves

    // ---- prefetch next tile into registers (pointer-increment addressing) ----
    #pragma unroll
    for (int u = 0; u < 16; u++) kpre.e[u] = (_Float16)kPre[u];
    #pragma unroll
    for (int u = 0; u < 16; u++) vpre.e[u] = (_Float16)vPre[(size_t)u * rowStride];

    // ---- wave-uniform tile classification; SGPR conditions => s_cbranch ----
    const int fullyMasked = __builtin_amdgcn_readfirstlane(causal && (kbase > qmax));
    if (!fullyMasked) {
      const int needMask = __builtin_amdgcn_readfirstlane(causal && (kbase + 31 > qmin));
      if (needMask)
        tileCompute(kbase, cur, std::integral_constant<bool, true>{});
      else
        tileCompute(kbase, cur, std::integral_constant<bool, false>{});
    }

    // ---- stage prefetched tile to the other buffer (readers done pre-barrier) ----
    {
      uint4* kd = (uint4*)&kTile[cur ^ 1][ksrow][ksfb];
      kd[0] = kpre.q[0]; kd[1] = kpre.q[1];
      uint4* vd = (uint4*)&vT[cur ^ 1][vsft][vskb];
      vd[0] = vpre.q[0]; vd[1] = vpre.q[1];
    }
    if (kt + 1 < ktmax) { kPre += tileStep; vPre += tileStep; }   // clamped advance
  }

  // ---- epilogue: divide by softmax denominator, scatter to dilated slots ----
  #pragma unroll
  for (int r = 0; r < 8; r++) {
    float inv = 1.0f / lrow[r];
    int qm = qmin + r + 8 * c;
    float* op = Out + base + (size_t)qm * rowStride + n16;
    #pragma unroll
    for (int f = 0; f < 4; f++) op[16 * f] = o[f].e[r] * inv;
  }
}

__global__ void da_colsum(const float* __restrict__ out, float* __restrict__ S) {
  __shared__ float red[256];
  int b = blockIdx.x / NH, h = blockIdx.x % NH;
  int t = threadIdx.x;
  int d = t & 63, chunk = t >> 6;
  float acc = 0.0f;
  for (int s_ = chunk; s_ < SEQ; s_ += 4)
    acc += out[((size_t)(b * SEQ + s_) * NH + h) * HD + d];
  red[t] = acc;
  __syncthreads();
  if (chunk == 0)
    S[(b * NH + h) * HD + d] = red[d] + red[64 + d] + red[128 + d] + red[192 + d];
}

__global__ void da_divide(float* __restrict__ out, const float* __restrict__ S) {
  int i = blockIdx.x * 256 + threadIdx.x;
  if (i >= NOUT) return;
  int d = i & 63;
  int rem = i >> 6;                 // (b*SEQ + s)*NH + h
  int h = rem % NH;
  int b = rem / (SEQ * NH);
  out[i] = out[i] / S[(b * NH + h) * HD + d];
}

extern "C" void kernel_launch(void* const* d_in, const int* in_sizes, int n_in,
                              void* d_out, int out_size, void* d_ws, size_t ws_size,
                              hipStream_t stream) {
  const float* Q = (const float*)d_in[0];
  const float* K = (const float*)d_in[1];
  const float* V = (const float*)d_in[2];
  const int* causal = (const int*)d_in[3];
  float* out  = (float*)d_out;
  float* colS = (float*)d_ws;   // 1536 floats

  da_zero<<<(NOUT + 255) / 256, 256, 0, stream>>>(out);
  da_flash<<<dim3(32, 56), 128, 0, stream>>>(Q, K, V, causal, out);
  da_colsum<<<2 * NH, 256, 0, stream>>>(out, colS);
  da_divide<<<(NOUT + 255) / 256, 256, 0, stream>>>(out, colS);
}
